// CausalNet_62766652064278
// MI455X (gfx1250) — matrix operand order repaired and checked by reading, not compiled, not run on hardware
//
#include <hip/hip_runtime.h>
#include <hip/hip_bf16.h>
#include <math.h>

// ---------------------------------------------------------------------------
// CausalNet forward for MI455X (gfx1250, wave32, WMMA).
// All heavy math runs through v_wmma_f32_16x16x32_bf16 with f32 accumulation.
// Weights are pre-converted (once per call) to bf16 *transposed* [N,K] so the
// WMMA B tiles are contiguous along K. Activations stay fp32 (LN / residual /
// softmax fidelity) and are converted to bf16 while staging A tiles into LDS.
// GEMM main loop is software-pipelined: double-buffered LDS, next tile's
// global loads issued before the current tile's WMMA block so s_wait_loadcnt
// lands after the matrix math. Workspace requirement: ~0.95 GB.
// ---------------------------------------------------------------------------

typedef __attribute__((ext_vector_type(16))) __bf16 v16bf;
typedef __attribute__((ext_vector_type(8)))  float  v8f;

#define R_TOT  32768          // B*N token rows per stream
#define TILE_M 128
#define TILE_N 128
#define TILE_K 32
#define LDSK   40             // 32 + 8 pad -> 80B row stride, conflict-free & 16B aligned

// ---------------------------------------------------------------------------
// Weight convert + transpose: fp32 [L,K,N] -> bf16 [L,N,K]
// ---------------------------------------------------------------------------
__global__ void cvt_wT_kernel(const float* __restrict__ src, __bf16* __restrict__ dst,
                              int K, int N, long long total) {
    long long i = (long long)blockIdx.x * blockDim.x + threadIdx.x;
    if (i >= total) return;
    long long kn = (long long)K * N;
    long long l  = i / kn;
    long long r  = i - l * kn;
    int k = (int)(r / N);
    int n = (int)(r - (long long)k * N);
    dst[l * kn + (long long)n * K + k] = (__bf16)src[i];
}

// ---------------------------------------------------------------------------
// LayerNorm over last dim (512). One wave per row, shfl_xor reduction.
// ---------------------------------------------------------------------------
__global__ __launch_bounds__(256) void ln_kernel(const float* __restrict__ x,
                                                 const float* __restrict__ w,
                                                 const float* __restrict__ b,
                                                 float* __restrict__ out) {
    int row  = blockIdx.x * 8 + (threadIdx.x >> 5);
    int lane = threadIdx.x & 31;
    const float* xr = x + (size_t)row * 512;
    float4 v[4];
    float s = 0.f, s2 = 0.f;
#pragma unroll
    for (int i = 0; i < 4; ++i) {
        v[i] = *(const float4*)(xr + i * 128 + lane * 4);
        s  += v[i].x + v[i].y + v[i].z + v[i].w;
        s2 += v[i].x * v[i].x + v[i].y * v[i].y + v[i].z * v[i].z + v[i].w * v[i].w;
    }
#pragma unroll
    for (int o = 16; o > 0; o >>= 1) { s += __shfl_xor(s, o); s2 += __shfl_xor(s2, o); }
    float mu   = s * (1.f / 512.f);
    float var  = s2 * (1.f / 512.f) - mu * mu;
    float rstd = rsqrtf(var + 1e-5f);
    float* orow = out + (size_t)row * 512;
#pragma unroll
    for (int i = 0; i < 4; ++i) {
        int c = i * 128 + lane * 4;
        float4 o;
        o.x = (v[i].x - mu) * rstd * w[c + 0] + b[c + 0];
        o.y = (v[i].y - mu) * rstd * w[c + 1] + b[c + 1];
        o.z = (v[i].z - mu) * rstd * w[c + 2] + b[c + 2];
        o.w = (v[i].w - mu) * rstd * w[c + 3] + b[c + 3];
        *(float4*)(orow + c) = o;
    }
}

// ---------------------------------------------------------------------------
// WMMA GEMM:  C[M,N] = A_f32[M,K] * WT_bf16[N,K]^T + bias  (+res | gelu)
// 128x128 block, 8 waves, each wave: 32x64 = 2x4 wmma 16x16x32 tiles.
// Double-buffered LDS; next-tile global loads overlap current-tile WMMA.
// EPI: 0 = bias, 1 = bias + residual, 2 = bias + exact GELU
// ---------------------------------------------------------------------------
__device__ __forceinline__ float gelu_exact(float x) {
    return 0.5f * x * (1.0f + erff(x * 0.70710678118654752f));
}

template <int EPI>
__global__ __launch_bounds__(256) void gemm_bf16_wmma(
    const float* __restrict__ A, const __bf16* __restrict__ WT,
    const float* __restrict__ bias, const float* __restrict__ Res,
    float* __restrict__ C, int M, int N, int K) {
    __shared__ __bf16 As[2][TILE_M][LDSK];
    __shared__ __bf16 Bs[2][TILE_N][LDSK];

    const int tid   = threadIdx.x;
    const int lane  = tid & 31;
    const int wave  = tid >> 5;
    const int waveM = wave & 3;   // 4 waves along M (4*32 = 128)
    const int waveN = wave >> 2;  // 2 waves along N (2*64 = 128)
    const int m0 = blockIdx.y * TILE_M;
    const int n0 = blockIdx.x * TILE_N;

    v8f acc[2][4];
#pragma unroll
    for (int mt = 0; mt < 2; ++mt)
#pragma unroll
        for (int nt = 0; nt < 4; ++nt)
#pragma unroll
            for (int e = 0; e < 8; ++e) acc[mt][nt][e] = 0.0f;

    const int sr = tid >> 1;   // staging row 0..127
    const int sh = tid & 1;    // staging K-half (16 elements)
    const int l  = lane & 15;
    const int hs = lane >> 4;

    union U { uint4 u[2]; v16bf v; };

    float4 aReg[4];            // next A sub-tile (fp32, 16 elems/thread)
    uint4  bReg[2];            // next B sub-tile (bf16, 16 elems/thread)

    const float* aBase = A  + (size_t)(m0 + sr) * K + sh * 16;
    const __bf16* bBase = WT + (size_t)(n0 + sr) * K + sh * 16;

    auto gload = [&](int k0) {
        const float4* ap = (const float4*)(aBase + k0);
#pragma unroll
        for (int q = 0; q < 4; ++q) aReg[q] = ap[q];
        const uint4* bp = (const uint4*)(bBase + k0);
        bReg[0] = bp[0];
        bReg[1] = bp[1];
    };
    auto lstore = [&](int buf) {
#pragma unroll
        for (int q = 0; q < 4; ++q) {
            int cb = sh * 16 + q * 4;
            As[buf][sr][cb + 0] = (__bf16)aReg[q].x;
            As[buf][sr][cb + 1] = (__bf16)aReg[q].y;
            As[buf][sr][cb + 2] = (__bf16)aReg[q].z;
            As[buf][sr][cb + 3] = (__bf16)aReg[q].w;
        }
        *(uint4*)&Bs[buf][sr][sh * 16]     = bReg[0];
        *(uint4*)&Bs[buf][sr][sh * 16 + 8] = bReg[1];
    };

    // prologue: stage tile 0
    gload(0);
    lstore(0);
    __syncthreads();

    const int nk = K / TILE_K;
    for (int kt = 0; kt < nk; ++kt) {
        const int cur = kt & 1;
        // issue next tile's global loads before the WMMA block (latency overlap)
        if (kt + 1 < nk) gload((kt + 1) * TILE_K);

        // A fragments (ISA 7.12.2: lane<16 K{0..7,16..23}, lane>=16 K{8..15,24..31})
        v16bf afrag[2], bfrag[4];
#pragma unroll
        for (int mt = 0; mt < 2; ++mt) {
            int row = waveM * 32 + mt * 16 + l;
            U t;
            t.u[0] = *(const uint4*)&As[cur][row][hs * 8];
            t.u[1] = *(const uint4*)&As[cur][row][hs * 8 + 16];
            afrag[mt] = t.v;
        }
        // B fragments: lane n holds column n, K-half chosen by lane[4]
#pragma unroll
        for (int nt = 0; nt < 4; ++nt) {
            int col = waveN * 64 + nt * 16 + l;
            U t;
            t.u[0] = *(const uint4*)&Bs[cur][col][hs * 8];
            t.u[1] = *(const uint4*)&Bs[cur][col][hs * 8 + 16];
            bfrag[nt] = t.v;
        }
#pragma unroll
        for (int mt = 0; mt < 2; ++mt)
#pragma unroll
            for (int nt = 0; nt < 4; ++nt)
                acc[mt][nt] = __builtin_amdgcn_wmma_f32_16x16x32_bf16(
                    false, afrag[mt], false, bfrag[nt], (short)0, acc[mt][nt],
                    false, false);

        // store next tile into the other buffer after the math is issued
        if (kt + 1 < nk) lstore(cur ^ 1);
        __syncthreads();
    }

    // Epilogue: C/D layout — VGPR e: M = e + (lane>=16 ? 8 : 0), N = lane&15
#pragma unroll
    for (int mt = 0; mt < 2; ++mt)
#pragma unroll
        for (int nt = 0; nt < 4; ++nt)
#pragma unroll
            for (int e = 0; e < 8; ++e) {
                int row = m0 + waveM * 32 + mt * 16 + e + hs * 8;
                int col = n0 + waveN * 64 + nt * 16 + l;
                float x = acc[mt][nt][e] + bias[col];
                if (EPI == 1) x += Res[(size_t)row * N + col];
                if (EPI == 2) x = gelu_exact(x);
                C[(size_t)row * N + col] = x;
            }
}

// ---------------------------------------------------------------------------
// Spatial attention: 4-token masked/biased attention, 1 thread per (b,h,i)
// ---------------------------------------------------------------------------
__global__ __launch_bounds__(256) void spatial_attn_kernel(
    const float* __restrict__ Q, const float* __restrict__ KV,
    float* __restrict__ AO) {
    const float MM[4][4] = {{1,1,1,0},{1,1,0,1},{1,0,1,1},{0,1,1,1}};
    const float PB[4][4] = {{0,0,0,-1.6f},{0,0,-1.2f,0},{0,-0.8f,0,0},{-0.4f,0,0,0}};
    long long t = (long long)blockIdx.x * 256 + threadIdx.x;   // (b*8+h)*4+i
    int i = (int)(t & 3);
    int h = (int)((t >> 2) & 7);
    long long bq = t >> 5;
    const float* q = Q + ((size_t)(bq * 4 + i)) * 512 + h * 64;
    float sc[4];
#pragma unroll
    for (int j = 0; j < 4; ++j) {
        const float* k = KV + ((size_t)(bq * 4 + j)) * 1024 + h * 64;
        float d = 0.f;
        for (int c = 0; c < 64; ++c) d += q[c] * k[c];
        sc[j] = d * 0.125f * MM[i][j] + PB[i][j];
    }
    float mx = fmaxf(fmaxf(sc[0], sc[1]), fmaxf(sc[2], sc[3]));
    float e[4], se = 0.f;
#pragma unroll
    for (int j = 0; j < 4; ++j) { e[j] = expf(sc[j] - mx); se += e[j]; }
    float inv = 1.f / se;
    float wgt[4];
#pragma unroll
    for (int j = 0; j < 4; ++j) wgt[j] = e[j] * inv * MM[i][j];
    float* o = AO + ((size_t)(bq * 4 + i)) * 512 + h * 64;
    for (int c = 0; c < 64; ++c) {
        float a = 0.f;
#pragma unroll
        for (int j = 0; j < 4; ++j)
            a += wgt[j] * KV[((size_t)(bq * 4 + j)) * 1024 + 512 + h * 64 + c];
        o[c] = a;
    }
}

// ---------------------------------------------------------------------------
// Temporal attention: 2-key attention per token, 1 thread per (b,h,n)
// ---------------------------------------------------------------------------
__global__ __launch_bounds__(256) void temporal_attn_kernel(
    const float* __restrict__ Q, const float* __restrict__ KV1,
    const float* __restrict__ KV2, float* __restrict__ AO) {
    long long t = (long long)blockIdx.x * 256 + threadIdx.x;
    int n = (int)(t & 3);
    int h = (int)((t >> 2) & 7);
    long long bq = t >> 5;
    size_t row = (size_t)(bq * 4 + n);
    const float* q  = Q   + row * 512  + h * 64;
    const float* k  = KV1 + row * 1024 + h * 64;
    const float* ks = KV2 + row * 1024 + h * 64;
    float s0 = 0.f, s1 = 0.f;
    for (int c = 0; c < 64; ++c) { s0 += q[c] * k[c]; s1 += q[c] * ks[c]; }
    s0 *= 0.125f; s1 *= 0.125f;
    float mx = fmaxf(s0, s1);
    float e0 = expf(s0 - mx), e1 = expf(s1 - mx);
    float inv = 1.f / (e0 + e1);
    e0 *= inv; e1 *= inv;
    const float* v  = KV1 + row * 1024 + 512 + h * 64;
    const float* vs = KV2 + row * 1024 + 512 + h * 64;
    float* o = AO + row * 512 + h * 64;
    for (int c = 0; c < 64; ++c) o[c] = e0 * v[c] + e1 * vs[c];
}

// ---------------------------------------------------------------------------
// CRM attention: unmasked 4-token attention + (y_for + y_back) residual
// ---------------------------------------------------------------------------
__global__ __launch_bounds__(256) void crm_attn_kernel(
    const float* __restrict__ Q, const float* __restrict__ Kb,
    const float* __restrict__ Vb, const float* __restrict__ Yf,
    const float* __restrict__ Yb, float* __restrict__ AO) {
    long long t = (long long)blockIdx.x * 256 + threadIdx.x;
    int i = (int)(t & 3);
    int h = (int)((t >> 2) & 7);
    long long bq = t >> 5;
    const float* q = Q + ((size_t)(bq * 4 + i)) * 512 + h * 64;
    float sc[4];
#pragma unroll
    for (int j = 0; j < 4; ++j) {
        const float* k = Kb + ((size_t)(bq * 4 + j)) * 512 + h * 64;
        float d = 0.f;
        for (int c = 0; c < 64; ++c) d += q[c] * k[c];
        sc[j] = d * 0.125f;
    }
    float mx = fmaxf(fmaxf(sc[0], sc[1]), fmaxf(sc[2], sc[3]));
    float e[4], se = 0.f;
#pragma unroll
    for (int j = 0; j < 4; ++j) { e[j] = expf(sc[j] - mx); se += e[j]; }
    float inv = 1.f / se;
    size_t rowi = (size_t)(bq * 4 + i);
    float* o = AO + rowi * 512 + h * 64;
    for (int c = 0; c < 64; ++c) {
        float a = 0.f;
#pragma unroll
        for (int j = 0; j < 4; ++j)
            a += e[j] * inv * Vb[((size_t)(bq * 4 + j)) * 512 + h * 64 + c];
        o[c] = a + Yf[rowi * 512 + h * 64 + c] + Yb[rowi * 512 + h * 64 + c];
    }
}

__global__ void add_kernel(const float* __restrict__ a, const float* __restrict__ b,
                           float* __restrict__ c, long long n) {
    long long i = (long long)blockIdx.x * 256 + threadIdx.x;
    if (i < n) c[i] = a[i] + b[i];
}

// out[b, slot, n, d]: slot0 = y_long, slot1 = y1f, slot2 = y1b
__global__ void gather_kernel(const float* __restrict__ YL, const float* __restrict__ Y1F,
                              const float* __restrict__ Y1B, float* __restrict__ out,
                              long long total) {
    long long i = (long long)blockIdx.x * 256 + threadIdx.x;
    if (i >= total) return;
    long long b    = i / 6144;            // 3*4*512
    long long r    = i - b * 6144;
    int       slot = (int)(r / 2048);
    long long nd   = r - (long long)slot * 2048;
    const float* src = (slot == 0) ? YL : ((slot == 1) ? Y1F : Y1B);
    out[i] = src[b * 2048 + nd];
}

// ---------------------------------------------------------------------------
// Host side
// ---------------------------------------------------------------------------
static void launch_gemm(int epi, const float* A, const __bf16* WT, const float* bias,
                        const float* Res, float* C, int M, int N, int K, hipStream_t s) {
    dim3 g(N / TILE_N, M / TILE_M), b(256);
    if (epi == 0)      gemm_bf16_wmma<0><<<g, b, 0, s>>>(A, WT, bias, Res, C, M, N, K);
    else if (epi == 1) gemm_bf16_wmma<1><<<g, b, 0, s>>>(A, WT, bias, Res, C, M, N, K);
    else               gemm_bf16_wmma<2><<<g, b, 0, s>>>(A, WT, bias, Res, C, M, N, K);
}

static void launch_cvt(const float* src, __bf16* dst, int K, int N, hipStream_t s) {
    long long total = 4ll * K * N;
    cvt_wT_kernel<<<(unsigned)((total + 255) / 256), 256, 0, s>>>(src, dst, K, N, total);
}

extern "C" void kernel_launch(void* const* d_in, const int* in_sizes, int n_in,
                              void* d_out, int out_size, void* d_ws, size_t ws_size,
                              hipStream_t stream) {
    // ---- inputs (setup_inputs order) ----
    const float* x1      = (const float*)d_in[0];
    const float* x2      = (const float*)d_in[1];
    const float* ln_w    = (const float*)d_in[2];
    const float* ln_b    = (const float*)d_in[3];
    const float* st_wq   = (const float*)d_in[4];
    const float* st_bq   = (const float*)d_in[5];
    const float* st_wkv  = (const float*)d_in[6];
    const float* st_bkv  = (const float*)d_in[7];
    const float* st_wo   = (const float*)d_in[8];
    const float* st_bo   = (const float*)d_in[9];
    const float* st_twq  = (const float*)d_in[10];
    const float* st_tbq  = (const float*)d_in[11];
    const float* st_twkv = (const float*)d_in[12];
    const float* st_tbkv = (const float*)d_in[13];
    const float* st_two  = (const float*)d_in[14];
    const float* st_tbo  = (const float*)d_in[15];
    const float* ff_w1   = (const float*)d_in[16];
    const float* ff_b1   = (const float*)d_in[17];
    const float* ff_w2   = (const float*)d_in[18];
    const float* ff_b2   = (const float*)d_in[19];
    const float* crm_wq  = (const float*)d_in[20];
    const float* crm_bq  = (const float*)d_in[21];
    const float* crm_wk  = (const float*)d_in[22];
    const float* crm_bk  = (const float*)d_in[23];
    const float* crm_wv  = (const float*)d_in[24];
    const float* crm_bv  = (const float*)d_in[25];
    const float* crm_wo  = (const float*)d_in[26];
    const float* crm_bo  = (const float*)d_in[27];

    // ---- workspace layout ----
    // bf16 transposed weights (element offsets)
    const size_t O_WQ   = 0;
    const size_t O_WKV  = O_WQ   + 4ull * 512 * 512;
    const size_t O_WO   = O_WKV  + 4ull * 512 * 1024;
    const size_t O_TWQ  = O_WO   + 4ull * 512 * 512;
    const size_t O_TWKV = O_TWQ  + 4ull * 512 * 512;
    const size_t O_TWO  = O_TWKV + 4ull * 512 * 1024;
    const size_t O_FW1  = O_TWO  + 4ull * 512 * 512;
    const size_t O_FW2  = O_FW1  + 4ull * 512 * 2048;
    const size_t O_CWQ  = O_FW2  + 4ull * 2048 * 512;
    const size_t O_CWK  = O_CWQ  + 4ull * 512 * 512;
    const size_t O_CWV  = O_CWK  + 4ull * 512 * 512;
    const size_t O_CWO  = O_CWV  + 4ull * 512 * 512;
    const size_t W_ELEMS = O_CWO + 4ull * 512 * 512;   // 20,971,520 elems = 40 MB

    char*   ws = (char*)d_ws;
    __bf16* WB = (__bf16*)ws;
    size_t  off = ((W_ELEMS * 2 + 255) / 256) * 256;
    auto take = [&](size_t bytes) {
        float* p = (float*)(ws + off);
        off += ((bytes + 255) / 256) * 256;
        return p;
    };
    const size_t SZ512  = (size_t)R_TOT * 512  * 4;   // 64 MB
    const size_t SZ1024 = (size_t)R_TOT * 1024 * 4;   // 128 MB
    const size_t SZ2048 = (size_t)R_TOT * 2048 * 4;   // 256 MB
    float* Y[4];                                       // y1f, y2f, y1b, y2b
    for (int i = 0; i < 4; ++i) Y[i] = take(SZ512);
    float* X1  = take(SZ512);
    float* X2  = take(SZ512);
    float* Qb  = take(SZ512);
    float* KVb = take(SZ1024);
    float* Hb  = take(SZ2048);    // doubles as second temporal KV and MLP hidden
    float* AOb = take(SZ512);
    (void)ws_size; (void)in_sizes; (void)n_in; (void)out_size;

    // ---- 1. convert + transpose all weights to bf16 [N,K] (L2-resident) ----
    launch_cvt(st_wq,   WB + O_WQ,   512,  512,  stream);
    launch_cvt(st_wkv,  WB + O_WKV,  512,  1024, stream);
    launch_cvt(st_wo,   WB + O_WO,   512,  512,  stream);
    launch_cvt(st_twq,  WB + O_TWQ,  512,  512,  stream);
    launch_cvt(st_twkv, WB + O_TWKV, 512,  1024, stream);
    launch_cvt(st_two,  WB + O_TWO,  512,  512,  stream);
    launch_cvt(ff_w1,   WB + O_FW1,  512,  2048, stream);
    launch_cvt(ff_w2,   WB + O_FW2,  2048, 512,  stream);
    launch_cvt(crm_wq,  WB + O_CWQ,  512,  512,  stream);
    launch_cvt(crm_wk,  WB + O_CWK,  512,  512,  stream);
    launch_cvt(crm_wv,  WB + O_CWV,  512,  512,  stream);
    launch_cvt(crm_wo,  WB + O_CWO,  512,  512,  stream);

    // ---- 2. seed streams: y1f=x1, y2f=x2, y1b=x2, y2b=x1 ----
    hipMemcpyAsync(Y[0], x1, SZ512, hipMemcpyDeviceToDevice, stream);
    hipMemcpyAsync(Y[1], x2, SZ512, hipMemcpyDeviceToDevice, stream);
    hipMemcpyAsync(Y[2], x2, SZ512, hipMemcpyDeviceToDevice, stream);
    hipMemcpyAsync(Y[3], x1, SZ512, hipMemcpyDeviceToDevice, stream);

    const unsigned ATT_G = (unsigned)((8192ll * 8 * 4) / 256);   // 1024 blocks
    const unsigned LN_G  = R_TOT / 8;

    // ---- 3. layers ----
    for (int lyr = 0; lyr < 4; ++lyr) {
        size_t lb = (size_t)lyr;
        const __bf16* wqT   = WB + O_WQ   + lb * 512 * 512;
        const __bf16* wkvT  = WB + O_WKV  + lb * 512 * 1024;
        const __bf16* woT   = WB + O_WO   + lb * 512 * 512;
        const __bf16* twqT  = WB + O_TWQ  + lb * 512 * 512;
        const __bf16* twkvT = WB + O_TWKV + lb * 512 * 1024;
        const __bf16* twoT  = WB + O_TWO  + lb * 512 * 512;
        const __bf16* fw1T  = WB + O_FW1  + lb * 512 * 2048;
        const __bf16* fw2T  = WB + O_FW2  + lb * 2048 * 512;
        const float* lw  = ln_w    + lb * 512;
        const float* lbv = ln_b    + lb * 512;
        const float* bq  = st_bq   + lb * 512;
        const float* bkv = st_bkv  + lb * 1024;
        const float* bo  = st_bo   + lb * 512;
        const float* tbq = st_tbq  + lb * 512;
        const float* tbkv= st_tbkv + lb * 1024;
        const float* tbo = st_tbo  + lb * 512;
        const float* fb1 = ff_b1   + lb * 2048;
        const float* fb2 = ff_b2   + lb * 512;

        for (int pr = 0; pr < 2; ++pr) {
            float* ya = Y[pr * 2 + 0];
            float* yb = Y[pr * 2 + 1];
            // LN
            ln_kernel<<<LN_G, 256, 0, stream>>>(ya, lw, lbv, X1);
            ln_kernel<<<LN_G, 256, 0, stream>>>(yb, lw, lbv, X2);
            // spatial(x1) + x1  -> ya
            launch_gemm(0, X1, wqT,  bq,  nullptr, Qb,  R_TOT, 512,  512, stream);
            launch_gemm(0, X1, wkvT, bkv, nullptr, KVb, R_TOT, 1024, 512, stream);
            spatial_attn_kernel<<<ATT_G, 256, 0, stream>>>(Qb, KVb, AOb);
            launch_gemm(1, AOb, woT, bo, X1, ya, R_TOT, 512, 512, stream);
            // spatial(x2) + x2  -> yb
            launch_gemm(0, X2, wqT,  bq,  nullptr, Qb,  R_TOT, 512,  512, stream);
            launch_gemm(0, X2, wkvT, bkv, nullptr, KVb, R_TOT, 1024, 512, stream);
            spatial_attn_kernel<<<ATT_G, 256, 0, stream>>>(Qb, KVb, AOb);
            launch_gemm(1, AOb, woT, bo, X2, yb, R_TOT, 512, 512, stream);
            // temporal(ya, yb) + yb -> yb   (q from yb; k,v from ya; ks,vs from yb)
            launch_gemm(0, yb, twqT,  tbq,  nullptr, Qb,  R_TOT, 512,  512, stream);
            launch_gemm(0, ya, twkvT, tbkv, nullptr, KVb, R_TOT, 1024, 512, stream);
            launch_gemm(0, yb, twkvT, tbkv, nullptr, Hb,  R_TOT, 1024, 512, stream);
            temporal_attn_kernel<<<ATT_G, 256, 0, stream>>>(Qb, KVb, Hb, AOb);
            launch_gemm(1, AOb, twoT, tbo, yb, yb, R_TOT, 512, 512, stream);
            // FF(ya) + ya ; FF(yb) + yb   (Hb reused as 32768x2048 hidden)
            launch_gemm(2, ya, fw1T, fb1, nullptr, Hb, R_TOT, 2048, 512, stream);
            launch_gemm(1, Hb, fw2T, fb2, ya, ya, R_TOT, 512, 2048, stream);
            launch_gemm(2, yb, fw1T, fb1, nullptr, Hb, R_TOT, 2048, 512, stream);
            launch_gemm(1, Hb, fw2T, fb2, yb, yb, R_TOT, 512, 2048, stream);
        }
    }

    // ---- 4. CRM with layer-3 params: y_for = y2f (Y[1]), y_back = y2b (Y[3]) ----
    {
        const size_t lb = 3;
        const __bf16* cwqT = WB + O_CWQ + lb * 512 * 512;
        const __bf16* cwkT = WB + O_CWK + lb * 512 * 512;
        const __bf16* cwvT = WB + O_CWV + lb * 512 * 512;
        const __bf16* cwoT = WB + O_CWO + lb * 512 * 512;
        const float* cbq = crm_bq + lb * 512;
        const float* cbk = crm_bk + lb * 512;
        const float* cbv = crm_bv + lb * 512;
        const float* cbo = crm_bo + lb * 512;
        long long nel = (long long)R_TOT * 512;
        add_kernel<<<(unsigned)((nel + 255) / 256), 256, 0, stream>>>(Y[1], Y[3], X2, nel);
        launch_gemm(0, Y[1], cwqT, cbq, nullptr, Qb,  R_TOT, 512, 512, stream);
        launch_gemm(0, Y[3], cwkT, cbk, nullptr, KVb, R_TOT, 512, 512, stream);
        launch_gemm(0, X2,   cwvT, cbv, nullptr, Hb,  R_TOT, 512, 512, stream);
        crm_attn_kernel<<<ATT_G, 256, 0, stream>>>(Qb, KVb, Hb, Y[1], Y[3], AOb);
        launch_gemm(0, AOb, cwoT, cbo, nullptr, X1, R_TOT, 512, 512, stream);  // y_long
    }

    // ---- 5. assemble output [B, 3, 4, 512] ----
    long long total = 8192ll * 3 * 4 * 512;
    gather_kernel<<<(unsigned)((total + 255) / 256), 256, 0, stream>>>(
        X1, Y[0], Y[2], (float*)d_out, total);
}
